// FastCMIF_51196010168323
// MI455X (gfx1250) — 8-bit matrix operand order repaired and checked by reading, compiled, not run on hardware
//
#include <hip/hip_runtime.h>
#include <stdint.h>

typedef __attribute__((ext_vector_type(16))) int   v16i;
typedef __attribute__((ext_vector_type(8)))  float v8f;

#define IMW    256
#define IMN    (IMW * IMW)
#define TW     32
#define TN     (TW * TW)
#define NBATCH 2
#define STEPS  16
#define ROWW   18
#define PLANEW 580

__device__ __forceinline__ uint32_t onehot4(uint32_t w, uint32_t m4) {
  uint32_t x = w ^ m4;
  uint32_t t = (x | (x >> 4)) & 0x0F0F0F0Fu;
  t = (t | (t >> 2)) & 0x03030303u;
  t = (t | (t >> 1)) & 0x01010101u;
  return (t ^ 0x01010101u) * 0x38u;
}

__global__ __launch_bounds__(256) void mm_kernel(const float* __restrict__ im,
                                                 const float* __restrict__ tp,
                                                 float* __restrict__ mm) {
  __shared__ float smn[256], smx[256];
  const int t = threadIdx.x, blk = blockIdx.x;
  const float* src; int n;
  if (blk < NBATCH) { src = im + blk * IMN; n = IMN; }
  else              { src = tp + (blk - NBATCH) * TN; n = TN; }
  float mn = 3.4e38f, mx = -3.4e38f;
  for (int i = t; i < n; i += 256) { float v = src[i]; mn = fminf(mn, v); mx = fmaxf(mx, v); }
  smn[t] = mn; smx[t] = mx; __syncthreads();
  for (int s = 128; s > 0; s >>= 1) {
    if (t < s) { smn[t] = fminf(smn[t], smn[t + s]); smx[t] = fmaxf(smx[t], smx[t + s]); }
    __syncthreads();
  }
  if (t == 0) { mm[blk * 2] = smn[0]; mm[blk * 2 + 1] = smx[0]; }
}

__global__ __launch_bounds__(256) void bin_kernel(const float* __restrict__ im,
                                                  const float* __restrict__ tp,
                                                  const float* __restrict__ mm,
                                                  uint8_t* __restrict__ imBin,
                                                  uint8_t* __restrict__ tBin) {
  const int i = blockIdx.x * 256 + threadIdx.x;
  if (i < NBATCH * IMN) {
    const int b = i >> 16;
    const float mn = mm[b * 2], mx = mm[b * 2 + 1];
    const float tv = (im[i] - mn) / (mx - mn);
    int idx = (int)floorf(tv * 15.0f);
    idx = idx < 0 ? 0 : (idx > 15 ? 15 : idx);
    imBin[i] = (uint8_t)idx;
  } else if (i < NBATCH * IMN + NBATCH * TN) {
    const int j = i - NBATCH * IMN;
    const int b = j >> 10;
    const float mn = mm[2 * NBATCH + b * 2], mx = mm[2 * NBATCH + b * 2 + 1];
    const float tv = (tp[j] - mn) / (mx - mn);
    int idx = (int)floorf(tv * 15.0f);
    idx = idx < 0 ? 0 : (idx > 15 ? 15 : idx);
    tBin[j] = (uint8_t)idx;
  }
}

__global__ __launch_bounds__(256) void btab_kernel(const uint8_t* __restrict__ tBin,
                                                   uint32_t* __restrict__ btab) {
  const int i = blockIdx.x * 256 + threadIdx.x;
  const int v    = i & 15;
  const int s    = (i >> 4) & 15;
  const int lane = (i >> 8) & 31;
  const int off  = (i >> 13) & 3;
  const int b    = (i >> 15) & 1;
  const int hi = lane >> 4, col = lane & 15;
  const int row  = 2 * s + (v >> 3);
  const int word = 4 * ((v & 7) >> 1) + (v & 1) + 2 * hi;
  uint32_t wv = 0;
  #pragma unroll
  for (int j = 0; j < 4; ++j) {
    const int mx = 4 * word + j - off;
    uint32_t byte = 0;
    if (mx >= 0 && mx < TW && (int)tBin[b * TN + row * TW + mx] == col) byte = 0x38u;
    wv |= byte << (8 * j);
  }
  btab[i] = wv;
}

__global__ __launch_bounds__(256) void cmif_kernel(const uint8_t* __restrict__ imBin,
                                                   const uint32_t* __restrict__ btab,
                                                   float* __restrict__ out) {
  __shared__ uint32_t planes[16][PLANEW];
  __shared__ uint32_t raw[32][12];

  const int tid  = threadIdx.x;
  const int lane = tid & 31;
  const int wave = tid >> 5;
  const int y  = blockIdx.y;
  const int x0 = blockIdx.x * 8;
  const int x  = x0 + wave;
  const int b  = blockIdx.z;
  const int q0min = (x0 - 15) >> 2;

  for (int i = tid; i < 16 * PLANEW; i += 256) (&planes[0][0])[i] = 0u;

  {
    const int row = tid >> 3;
    const int r = y - 15 + row;
    const bool rok = (r >= 0) & (r < IMW);
    const uint8_t* rp = imBin + b * IMN + (rok ? r : 0) * IMW;
    for (int wa = tid & 7; wa < 11; wa += 8) {
      uint32_t wv = 0;
      #pragma unroll
      for (int j = 0; j < 4; ++j) {
        const int c = 4 * (q0min + wa) + j;
        const bool cok = (c >= 0) & (c < IMW);
        uint32_t vb = rp[cok ? c : 0];
        if (!(rok & cok)) vb = 0xFFu;
        wv |= vb << (8 * j);
      }
      raw[row][wa] = wv;
    }
  }
  __syncthreads();

  {
    const int bin = tid & 15;
    const uint32_t m4 = (uint32_t)bin * 0x01010101u;
    for (int k = tid >> 4; k < 32 * 11; k += 16) {
      const int row = k / 11, wa = k - row * 11;
      planes[bin][row * ROWW + wa] = onehot4(raw[row][wa], m4);
    }
  }
  __syncthreads();

  const int M  = lane & 15;
  const int hi = lane >> 4;
  const int d   = ((x - 15) >> 2) - q0min;
  const int off = (x - 15) & 3;
  const uint32_t* ap = &planes[0][0] + M * PLANEW + d + 2 * hi;
  const uint32_t* bp = btab + (((b * 4 + off) * 32 + lane) << 8);

  v8f acc = {};
  #pragma unroll
  for (int s = 0; s < STEPS; ++s) {
    v16i A;
    #pragma unroll
    for (int v = 0; v < 16; ++v) {
      const int row = 2 * s + (v >> 3);
      const int w0  = 4 * ((v & 7) >> 1) + (v & 1);
      A[v] = (int)ap[row * ROWW + w0];
    }
    const v16i B = *(const v16i*)(bp + s * 16);
    acc = __builtin_amdgcn_wmma_f32_16x16x128_fp8_fp8(A, B, (short)0, acc,
                                                       false,  false);
  }

  float c8[8], rs[8];
  float cs = 0.0f;
  #pragma unroll
  for (int r = 0; r < 8; ++r) { c8[r] = acc[r]; cs += c8[r]; }
  cs += __shfl_xor(cs, 16, 32);

  #pragma unroll
  for (int r = 0; r < 8; ++r) {
    float v = c8[r];
    v += __shfl_xor(v, 1, 32);
    v += __shfl_xor(v, 2, 32);
    v += __shfl_xor(v, 4, 32);
    v += __shfl_xor(v, 8, 32);
    rs[r] = v;
  }
  float Nv = cs;
  Nv += __shfl_xor(Nv, 1, 32);
  Nv += __shfl_xor(Nv, 2, 32);
  Nv += __shfl_xor(Nv, 4, 32);
  Nv += __shfl_xor(Nv, 8, 32);

  const float logN  = __logf(Nv);
  const float logcs = __logf(fmaxf(cs, 1.0f));

  float part = 0.0f;
  #pragma unroll
  for (int r = 0; r < 8; ++r) {
    const float c = c8[r];
    if (c > 0.5f)
      part += c * (__logf(c) + logN - __logf(rs[r]) - logcs);
  }
  part += __shfl_xor(part, 1, 32);
  part += __shfl_xor(part, 2, 32);
  part += __shfl_xor(part, 4, 32);
  part += __shfl_xor(part, 8, 32);
  part += __shfl_xor(part, 16, 32);

  if (lane == 0) out[b * IMN + y * IMW + x] = part / Nv;
}

extern "C" void kernel_launch(void* const* d_in, const int* in_sizes, int n_in,
                              void* d_out, int out_size, void* d_ws, size_t ws_size,
                              hipStream_t stream) {
  const float* im = (const float*)d_in[0];
  const float* tp = (const float*)d_in[1];
  float* out = (float*)d_out;

  char* ws = (char*)d_ws;
  uint32_t* btab  = (uint32_t*)ws;
  float*    mm    = (float*)(ws + 262144);
  uint8_t*  imBin = (uint8_t*)(ws + 262144 + 256);
  uint8_t*  tBin  = (uint8_t*)(ws + 262144 + 256 + NBATCH * IMN);

  mm_kernel<<<4, 256, 0, stream>>>(im, tp, mm);
  const int total = NBATCH * IMN + NBATCH * TN;
  bin_kernel<<<(total + 255) / 256, 256, 0, stream>>>(im, tp, mm, imBin, tBin);
  btab_kernel<<<256, 256, 0, stream>>>(tBin, btab);

  dim3 grid(IMW / 8, IMW, NBATCH);
  cmif_kernel<<<grid, 256, 0, stream>>>(imBin, btab, out);
}
